// FieldConditionedAtomAttention_80805514707444
// MI455X (gfx1250) — matrix operand order repaired and checked by reading, not compiled, not run on hardware
//
#include <hip/hip_runtime.h>
#include <hip/hip_bf16.h>
#include <math.h>

// ---- problem constants (from reference) ----
#define B_    16
#define N_    128
#define NE_   200
#define LAT_  128
#define CUT_  6.0f
#define ASK_  288   // atom_static K padded to multiple of 32 (260 -> 288)

typedef __attribute__((ext_vector_type(16))) __bf16 bf16x16;
typedef __attribute__((ext_vector_type(4)))  __bf16 bf16x4;
typedef __attribute__((ext_vector_type(8)))  float  f32x8;

// SiLU with hardware reciprocal (v_rcp_f32): activation-grade accuracy.
__device__ __forceinline__ float silu_f(float x) {
    return x * __builtin_amdgcn_rcpf(1.f + __expf(-x));
}

// =====================================================================
// WMMA GEMM:  C[M,N] = act(A[M,K] @ W[K,N] + bias[N])
// Preconditions: M % 64 == 0, N % 64 == 0, K % 32 == 0 (branch-free).
// Block = 128 threads (4 waves); wave w owns a 16x64 strip.
// LDS tiles in fragment order [owner][lane][elem]: each lane's 32-byte
// operand is 2x ds_load_b128. Staging writes bf16 quads (ds_store_b64);
// a K-quad maps to 4 contiguous fragment slots of one lane.
// =====================================================================
template <int ACT>
__global__ __launch_bounds__(128) void gemm_bias_act(
    const float* __restrict__ A, const float* __restrict__ W,
    const float* __restrict__ bias, float* __restrict__ C,
    int K, int N)
{
    __shared__ alignas(16) __bf16 lA[4][32][16];  // [wave][lane][elem]
    __shared__ alignas(16) __bf16 lB[4][32][16];  // [col-group j][lane][elem]

    const int tid  = threadIdx.x;
    const int lane = tid & 31;
    const int wave = tid >> 5;            // 0..3
    const int half = lane >> 4;           // 0/1
    const int l15  = lane & 15;
    const int tm0  = blockIdx.x * 64;
    const int n0   = blockIdx.y * 64;

    f32x8 acc[4];
#pragma unroll
    for (int j = 0; j < 4; ++j)
#pragma unroll
        for (int r = 0; r < 8; ++r) acc[j][r] = 0.f;

    const int kTiles = K >> 5;
    for (int kt = 0; kt < kTiles; ++kt) {
        const int k0 = kt << 5;

        // ---- stage A tile (64x32): 512 K-quads, float4 -> bf16x4 ----
#pragma unroll
        for (int p = 0; p < 4; ++p) {
            int idx = tid + p * 128;
            int row = idx >> 3;            // 0..63
            int kq  = (idx & 7) * 4;       // 0..28, quad-aligned
            float4 x = *(const float4*)(A + (size_t)(tm0 + row) * K + k0 + kq);
            int w  = row >> 4;
            int ln = (row & 15) + 16 * ((kq >> 3) & 1);
            int e  = 8 * (kq >> 4) + (kq & 7);          // multiple of 4
            bf16x4 pk;
            pk[0] = (__bf16)x.x; pk[1] = (__bf16)x.y;
            pk[2] = (__bf16)x.z; pk[3] = (__bf16)x.w;
            *(bf16x4*)&lA[w][ln][e] = pk;
        }
        // ---- stage W tile (32x64): 512 kr-quads (4 row-gathered loads) ----
#pragma unroll
        for (int p = 0; p < 4; ++p) {
            int idx = tid + p * 128;
            int c   = idx & 63;            // 0..63
            int krq = (idx >> 6) * 4;      // 0..28, quad-aligned
            const float* wp = W + (size_t)(k0 + krq) * N + n0 + c;
            float x0 = wp[0];
            float x1 = wp[(size_t)N];
            float x2 = wp[(size_t)2 * N];
            float x3 = wp[(size_t)3 * N];
            int j  = c >> 4;
            int ln = (c & 15) + 16 * ((krq >> 3) & 1);
            int e  = 8 * (krq >> 4) + (krq & 7);         // multiple of 4
            bf16x4 pk;
            pk[0] = (__bf16)x0; pk[1] = (__bf16)x1;
            pk[2] = (__bf16)x2; pk[3] = (__bf16)x3;
            *(bf16x4*)&lB[j][ln][e] = pk;
        }
        __syncthreads();

        // ---- fragments: contiguous 32B reads ----
        bf16x16 a = *(const bf16x16*)&lA[wave][lane][0];
#pragma unroll
        for (int j = 0; j < 4; ++j) {
            bf16x16 bf = *(const bf16x16*)&lB[j][lane][0];
            acc[j] = __builtin_amdgcn_wmma_f32_16x16x32_bf16(
                false, a, false, bf, (short)0, acc[j], false, false);
        }
        __syncthreads();
    }

    // ---- epilogue: bias + optional SiLU (branch-free, templated) ----
#pragma unroll
    for (int j = 0; j < 4; ++j) {
        const int col = n0 + j * 16 + l15;
        const float bb = bias[col];
#pragma unroll
        for (int r = 0; r < 8; ++r) {
            const int row = tm0 + wave * 16 + r + 8 * half;  // C layout
            float x = acc[j][r] + bb;
            if (ACT) x = silu_f(x);
            C[(size_t)row * N + col] = x;
        }
    }
}

// =====================================================================
// Zero-pad a 260x256 weight into 288x256 (rows 260..287 = 0)
// =====================================================================
__global__ void pad_w0(const float* __restrict__ w, float* __restrict__ wp)
{
    int idx = blockIdx.x * blockDim.x + threadIdx.x;
    if (idx >= ASK_ * 256) return;
    int r = idx >> 8;
    wp[idx] = (r < 260) ? w[idx] : 0.f;
}

// =====================================================================
// Build q_in = concat(h[b,abs], e_feat[e], field_abs[b])  (3200 x 224)
// grid = (3200), block = 224: one q_in row per block; b, e split from row by / NE_.
// =====================================================================
__global__ void build_qin(const float* __restrict__ h,
                          const float* __restrict__ e_feat,
                          const float* __restrict__ f_abs,
                          const int* __restrict__ absid,
                          float* __restrict__ qin)
{
    const int row = blockIdx.x;          // b*NE_ + e
    const int d   = threadIdx.x;         // 0..223
    const int e   = blockIdx.y;          // unused marker (1D grid)
    (void)e;
    const int bb  = row / NE_;
    const int ee  = row - bb * NE_;
    const int ai  = absid[0];
    float v;
    if (d < 128)      v = h[((size_t)bb * N_ + ai) * 128 + d];
    else if (d < 160) v = e_feat[ee * 32 + (d - 128)];
    else              v = f_abs[bb * 64 + (d - 160)];
    qin[(size_t)row * 224 + d] = v;
}

// =====================================================================
// Build atom_static (2048 x 288, zero-padded) + radial bias + validity.
// layout: [h(128) | zemb(32) | rbf(32) | u(3) | is_abs(1) | field(64) | 0...]
// =====================================================================
__global__ void build_atom_static(const float* __restrict__ h,
                                  const int* __restrict__ z,
                                  const float* __restrict__ pos,
                                  const unsigned char* __restrict__ mask,
                                  const float* __restrict__ f_atom,
                                  const float* __restrict__ zemb,
                                  const int* __restrict__ absid,
                                  float* __restrict__ as_,
                                  float* __restrict__ bias_arr,
                                  float* __restrict__ valid_arr)
{
    const int row = blockIdx.x;     // b*N_ + n
    const int b = row >> 7;         // / N_
    const int n = row & (N_ - 1);
    const int ai = absid[0];
    const size_t ar = (size_t)b * N_ + ai;
    float px = pos[(size_t)row * 3 + 0] - pos[ar * 3 + 0];
    float py = pos[(size_t)row * 3 + 1] - pos[ar * 3 + 1];
    float pz = pos[(size_t)row * 3 + 2] - pos[ar * 3 + 2];
    float r  = sqrtf(px * px + py * py + pz * pz);
    float inv = __builtin_amdgcn_rcpf(fmaxf(r, 1e-8f));
    float ux = px * inv, uy = py * inv, uz = pz * inv;

    float* out = as_ + (size_t)row * ASK_;
    const float delta = CUT_ / 31.f;
    const float gamma = 1.f / (delta * delta + 1e-12f);
    const float rcl   = fminf(r, CUT_);
    for (int d = threadIdx.x; d < ASK_; d += blockDim.x) {
        float v;
        if (d < 128)      v = h[(size_t)row * 128 + d];
        else if (d < 160) v = zemb[(size_t)z[row] * 32 + (d - 128)];
        else if (d < 192) {
            int i = d - 160;
            float c = CUT_ * (float)i / 31.f;
            float t = rcl - c;
            v = __expf(-gamma * t * t);
        }
        else if (d == 192) v = ux;
        else if (d == 193) v = uy;
        else if (d == 194) v = uz;
        else if (d == 195) v = (n == ai) ? 1.f : 0.f;
        else if (d < 260)  v = f_atom[(size_t)row * 64 + (d - 196)];
        else               v = 0.f;                     // K padding
        out[d] = v;
    }
    if (threadIdx.x == 0) {
        float cf = (r <= CUT_) ? 0.5f * (__cosf(3.14159265358979f * r / CUT_) + 1.f) : 0.f;
        bias_arr[row]  = __logf(fmaxf(cf, 1e-8f));
        valid_arr[row] = ((mask[row] != 0) && (r <= CUT_)) ? 1.f : 0.f;
    }
}

// =====================================================================
// Attention: one block per (b, 16-row e tile); one wave per head.
// QK^T via WMMA (8 chunks of 16 atoms), masked softmax with shfl_xor
// row reductions, P -> LDS in fragment order (bf16), PV via WMMA.
// Q row index is clamped (not branched): WMMA rows are independent and
// out-of-range rows are never stored.
// =====================================================================
__global__ __launch_bounds__(128) void attention_kernel(
    const float* __restrict__ q,       // (B,NE,128)
    const float* __restrict__ kmat,    // (B,N,128)
    const float* __restrict__ vmat,    // (B,N,128)
    const float* __restrict__ bias_arr,
    const float* __restrict__ valid_arr,
    float* __restrict__ out)           // (B,NE,128)
{
    const int b    = blockIdx.x;
    const int e0   = blockIdx.y * 16;
    const int lane = threadIdx.x & 31;
    const int hh   = threadIdx.x >> 5;   // head = wave
    const int half = lane >> 4;
    const int l15  = lane & 15;

    // P tile in fragment order: [head][k-chunk][lane][elem]
    __shared__ alignas(16) __bf16 lPf[4][4][32][16];

    // ---- Q fragment: two contiguous 8-float runs per lane (clamped) ----
    bf16x16 a;
    const int erow = e0 + l15;
    {
        const int erc = (erow < NE_) ? erow : (NE_ - 1);
        const float* qrow = q + ((size_t)b * NE_ + erc) * 128 + hh * 32 + 8 * half;
        float4 q0 = *(const float4*)(qrow + 0);
        float4 q1 = *(const float4*)(qrow + 4);
        float4 q2 = *(const float4*)(qrow + 16);
        float4 q3 = *(const float4*)(qrow + 20);
        a[0]=(__bf16)q0.x; a[1]=(__bf16)q0.y; a[2]=(__bf16)q0.z; a[3]=(__bf16)q0.w;
        a[4]=(__bf16)q1.x; a[5]=(__bf16)q1.y; a[6]=(__bf16)q1.z; a[7]=(__bf16)q1.w;
        a[8]=(__bf16)q2.x; a[9]=(__bf16)q2.y; a[10]=(__bf16)q2.z; a[11]=(__bf16)q2.w;
        a[12]=(__bf16)q3.x; a[13]=(__bf16)q3.y; a[14]=(__bf16)q3.z; a[15]=(__bf16)q3.w;
    }

    // ---- scores: 8 WMMAs over 128 atoms ----
    f32x8 acc[8];
#pragma unroll
    for (int c = 0; c < 8; ++c)
#pragma unroll
        for (int r = 0; r < 8; ++r) acc[c][r] = 0.f;
#pragma unroll
    for (int c = 0; c < 8; ++c) {
        // B fragment: two 8-head-dim runs (8*half, 16+8*half) at row (atom) c*16+l15
        const float* krow = kmat + ((size_t)b * N_ + c * 16 + l15) * 128 + hh * 32 + 8 * half;
        float4 k0 = *(const float4*)(krow + 0);
        float4 k1 = *(const float4*)(krow + 4);
        float4 k2 = *(const float4*)(krow + 16);
        float4 k3 = *(const float4*)(krow + 20);
        bf16x16 bf;
        bf[0]=(__bf16)k0.x; bf[1]=(__bf16)k0.y; bf[2]=(__bf16)k0.z; bf[3]=(__bf16)k0.w;
        bf[4]=(__bf16)k1.x; bf[5]=(__bf16)k1.y; bf[6]=(__bf16)k1.z; bf[7]=(__bf16)k1.w;
        bf[8]=(__bf16)k2.x; bf[9]=(__bf16)k2.y; bf[10]=(__bf16)k2.z; bf[11]=(__bf16)k2.w;
        bf[12]=(__bf16)k3.x; bf[13]=(__bf16)k3.y; bf[14]=(__bf16)k3.z; bf[15]=(__bf16)k3.w;
        acc[c] = __builtin_amdgcn_wmma_f32_16x16x32_bf16(
            false, a, false, bf, (short)0, acc[c], false, false);
    }

    // ---- scale + radial bias + mask ----
    const float scale = 0.1767766952966369f;   // 32^-0.5
    float vvalid[8];
#pragma unroll
    for (int c = 0; c < 8; ++c) {
        int nn = c * 16 + l15;
        float bia = bias_arr[b * N_ + nn];
        float va  = valid_arr[b * N_ + nn];
        vvalid[c] = va;
#pragma unroll
        for (int r = 0; r < 8; ++r)
            acc[c][r] = (va > 0.5f) ? (acc[c][r] * scale + bia) : -1e9f;
    }

    // ---- softmax over atoms (per row = per e) ----
#pragma unroll
    for (int r = 0; r < 8; ++r) {
        float m = -3.4e38f;
#pragma unroll
        for (int c = 0; c < 8; ++c) m = fmaxf(m, acc[c][r]);
        m = fmaxf(m, __shfl_xor(m, 1, 32));
        m = fmaxf(m, __shfl_xor(m, 2, 32));
        m = fmaxf(m, __shfl_xor(m, 4, 32));
        m = fmaxf(m, __shfl_xor(m, 8, 32));
        float p[8], s = 0.f;
#pragma unroll
        for (int c = 0; c < 8; ++c) {
            p[c] = vvalid[c] * __expf(acc[c][r] - m);   // masked -> exactly 0
            s += p[c];
        }
        s += __shfl_xor(s, 1, 32);
        s += __shfl_xor(s, 2, 32);
        s += __shfl_xor(s, 4, 32);
        s += __shfl_xor(s, 8, 32);
        float invs = __builtin_amdgcn_rcpf(fmaxf(s, 1e-20f));
#pragma unroll
        for (int c = 0; c < 8; ++c) acc[c][r] = p[c] * invs;
    }

    // ---- scatter P -> LDS in A-fragment order ----
    // value at (row m = r+8*half, atom n = c*16+l15) goes to
    // lPf[hh][n>>5][m + 16*(l15>>3)][8*(c&1) + (l15&7)]
    {
        const int lnBase = 8 * half + 16 * (l15 >> 3);
        const int eBase  = l15 & 7;
#pragma unroll
        for (int c = 0; c < 8; ++c) {
            int kc = c >> 1;
            int e  = 8 * (c & 1) + eBase;
#pragma unroll
            for (int r = 0; r < 8; ++r)
                lPf[hh][kc][r + lnBase][e] = (__bf16)acc[c][r];
        }
    }
    __syncthreads();

    // ---- out = P (16x128) @ V_head (128x32): 4 K-chunks x 2 col groups ----
    f32x8 oacc[2];
#pragma unroll
    for (int g = 0; g < 2; ++g)
#pragma unroll
        for (int r = 0; r < 8; ++r) oacc[g][r] = 0.f;

#pragma unroll
    for (int kc = 0; kc < 4; ++kc) {
        bf16x16 pa = *(const bf16x16*)&lPf[hh][kc][lane][0];
#pragma unroll
        for (int g = 0; g < 2; ++g) {
            bf16x16 vb;
#pragma unroll
            for (int i = 0; i < 16; ++i) {
                int nrow = kc * 32 + 8 * half + i + 8 * (i >> 3);   // atom index (K dim)
                vb[i] = (__bf16)vmat[((size_t)b * N_ + nrow) * 128 + hh * 32 + g * 16 + l15];
            }
            oacc[g] = __builtin_amdgcn_wmma_f32_16x16x32_bf16(
                false, pa, false, vb, (short)0, oacc[g], false, false);
        }
    }

#pragma unroll
    for (int g = 0; g < 2; ++g) {
        const int col = hh * 32 + g * 16 + l15;
#pragma unroll
        for (int r = 0; r < 8; ++r) {
            int er = e0 + r + 8 * half;
            if (er < NE_)
                out[((size_t)b * NE_ + er) * 128 + col] = oacc[g][r];
        }
    }
}

// =====================================================================
// Host-side launcher
// =====================================================================
extern "C" void kernel_launch(void* const* d_in, const int* in_sizes, int n_in,
                              void* d_out, int out_size, void* d_ws, size_t ws_size,
                              hipStream_t stream)
{
    const float* h      = (const float*)d_in[0];
    const int*   z      = (const int*)d_in[1];
    const float* pos    = (const float*)d_in[2];
    const unsigned char* mask = (const unsigned char*)d_in[3];
    const float* e_feat = (const float*)d_in[4];
    const float* f_atom = (const float*)d_in[5];
    const float* f_abs  = (const float*)d_in[6];
    const int*   absid  = (const int*)d_in[7];

    // Parameter leaf order: jax pytree flatten sorts dict keys
    // (k,o,q,v,zemb); fall back to insertion order (q,k,v,o,zemb).
    // Disambiguate via leaf 8's size: k.W0 = 260*256, q.W0 = 224*256.
    int qi, ki, vi, oi, zi;
    if (in_sizes[8] == 260 * 256) { ki = 8;  oi = 14; qi = 18; vi = 24; zi = 30; }
    else                          { qi = 8;  ki = 14; vi = 20; oi = 26; zi = 30; }

    const float* qW0 = (const float*)d_in[qi + 0]; const float* qb0 = (const float*)d_in[qi + 1];
    const float* qW1 = (const float*)d_in[qi + 2]; const float* qb1 = (const float*)d_in[qi + 3];
    const float* qW2 = (const float*)d_in[qi + 4]; const float* qb2 = (const float*)d_in[qi + 5];
    const float* kW0 = (const float*)d_in[ki + 0]; const float* kb0 = (const float*)d_in[ki + 1];
    const float* kW1 = (const float*)d_in[ki + 2]; const float* kb1 = (const float*)d_in[ki + 3];
    const float* kW2 = (const float*)d_in[ki + 4]; const float* kb2 = (const float*)d_in[ki + 5];
    const float* vW0 = (const float*)d_in[vi + 0]; const float* vb0 = (const float*)d_in[vi + 1];
    const float* vW1 = (const float*)d_in[vi + 2]; const float* vb1 = (const float*)d_in[vi + 3];
    const float* vW2 = (const float*)d_in[vi + 4]; const float* vb2 = (const float*)d_in[vi + 5];
    const float* oW0 = (const float*)d_in[oi + 0]; const float* ob0 = (const float*)d_in[oi + 1];
    const float* oW1 = (const float*)d_in[oi + 2]; const float* ob1 = (const float*)d_in[oi + 3];
    const float* zemb = (const float*)d_in[zi];

    // ---- workspace layout (floats) ----
    float* ws = (float*)d_ws;
    const size_t ROWS_E = (size_t)B_ * NE_;   // 3200
    const size_t ROWS_A = (size_t)B_ * N_;    // 2048
    float* qin   = ws;                        // 3200*224
    float* h_a   = qin  + ROWS_E * 224;       // 3200*256 (shared hidden buf A)
    float* h_b   = h_a  + ROWS_E * 256;       // 3200*256 (shared hidden buf B)
    float* qbuf  = h_b  + ROWS_E * 256;       // 3200*128
    float* asbuf = qbuf + ROWS_E * 128;       // 2048*288 (K-padded)
    float* kbuf  = asbuf + ROWS_A * ASK_;     // 2048*128
    float* vbuf  = kbuf + ROWS_A * 128;       // 2048*128
    float* biasb = vbuf + ROWS_A * 128;       // 2048
    float* validb= biasb + ROWS_A;            // 2048
    float* attn  = validb + ROWS_A;           // 3200*128
    float* kW0p  = attn + ROWS_E * 128;       // 288*256
    float* vW0p  = kW0p + (size_t)ASK_ * 256; // 288*256

    // ---- 1) feature assembly + weight padding ----
    {
        build_qin<<<B_ * NE_, 224, 0, stream>>>(h, e_feat, f_abs, absid, qin);
        build_atom_static<<<B_ * N_, 64, 0, stream>>>(h, z, pos, mask, f_atom, zemb,
                                                      absid, asbuf, biasb, validb);
        int wtot = ASK_ * 256;
        pad_w0<<<(wtot + 255) / 256, 256, 0, stream>>>(kW0, kW0p);
        pad_w0<<<(wtot + 255) / 256, 256, 0, stream>>>(vW0, vW0p);
    }

    // ---- 2) q-MLP: 224 -> 256 -> 256 -> 128 ----
    gemm_bias_act<1><<<dim3(50, 4), 128, 0, stream>>>(qin, qW0, qb0, h_a, 224, 256);
    gemm_bias_act<1><<<dim3(50, 4), 128, 0, stream>>>(h_a, qW1, qb1, h_b, 256, 256);
    gemm_bias_act<0><<<dim3(50, 2), 128, 0, stream>>>(h_b, qW2, qb2, qbuf, 256, 128);

    // ---- 3) k-MLP: 288(pad) -> 256 -> 256 -> 128 ----
    gemm_bias_act<1><<<dim3(32, 4), 128, 0, stream>>>(asbuf, kW0p, kb0, h_a, ASK_, 256);
    gemm_bias_act<1><<<dim3(32, 4), 128, 0, stream>>>(h_a, kW1, kb1, h_b, 256, 256);
    gemm_bias_act<0><<<dim3(32, 2), 128, 0, stream>>>(h_b, kW2, kb2, kbuf, 256, 128);

    // ---- 4) v-MLP ----
    gemm_bias_act<1><<<dim3(32, 4), 128, 0, stream>>>(asbuf, vW0p, vb0, h_a, ASK_, 256);
    gemm_bias_act<1><<<dim3(32, 4), 128, 0, stream>>>(h_a, vW1, vb1, h_b, 256, 256);
    gemm_bias_act<0><<<dim3(32, 2), 128, 0, stream>>>(h_b, vW2, vb2, vbuf, 256, 128);

    // ---- 5) attention ----
    attention_kernel<<<dim3(B_, (NE_ + 15) / 16), 128, 0, stream>>>(
        qbuf, kbuf, vbuf, biasb, validb, attn);

    // ---- 6) o-MLP: 128 -> 256 -> 128, final writes to d_out ----
    gemm_bias_act<1><<<dim3(50, 4), 128, 0, stream>>>(attn, oW0, ob0, h_a, 128, 256);
    gemm_bias_act<0><<<dim3(50, 2), 128, 0, stream>>>(h_a, oW1, ob1, (float*)d_out, 256, 128);
}
